// MultiAttention_25305947308756
// MI455X (gfx1250) — matrix operand order repaired and checked by reading, not compiled, not run on hardware
//
#include <hip/hip_runtime.h>
#include <math.h>

typedef __attribute__((ext_vector_type(16))) _Float16 v16h;
typedef __attribute__((ext_vector_type(8)))  float    v8f;
typedef __attribute__((ext_vector_type(4)))  float    v4f;

#define HOPN 32
#define DIM  64

// Branch-free tanh: native V_TANH_F32 if declared, else exp2/rcp fallback.
__device__ __forceinline__ float fast_tanh(float x) {
#if __has_builtin(__builtin_amdgcn_tanhf)
  return __builtin_amdgcn_tanhf(x);
#else
  // tanh(x) = (e^{2x}-1)/(e^{2x}+1); clamp so exp2 stays finite (saturates to +/-1).
  float xc  = fminf(fmaxf(x, -15.0f), 15.0f);
  float e2x = __builtin_amdgcn_exp2f(xc * 2.8853900817779268f);  // 2*log2(e)
  return (e2x - 1.0f) * __builtin_amdgcn_rcpf(e2x + 1.0f);
#endif
}

__device__ __forceinline__ float fast_exp_nonpos(float x) {
  // x <= 0 here (softmax after max subtraction): single v_exp_f32, no guards.
  return __builtin_amdgcn_exp2f(x * 1.4426950408889634f);
}

// Build a WMMA A-tile (16x32 f16) for this lane from a row-major fp32 row.
// A layout (16-bit, 16x32): lanes 0-15 & 16-31 both hold row M=lane&15;
// halves 0..7 -> K = kb..kb+7, halves 8..15 -> K = kb+16..kb+23,
// where kb = kc*32 + (lane>>4)*8.
__device__ __forceinline__ v16h load_a_tile(const float* __restrict__ rowp, int kb) {
  v4f r0 = *(const v4f*)(rowp + kb);
  v4f r1 = *(const v4f*)(rowp + kb + 4);
  v4f r2 = *(const v4f*)(rowp + kb + 16);
  v4f r3 = *(const v4f*)(rowp + kb + 20);
  v16h a;
  a[0]  = (_Float16)r0[0]; a[1]  = (_Float16)r0[1]; a[2]  = (_Float16)r0[2]; a[3]  = (_Float16)r0[3];
  a[4]  = (_Float16)r1[0]; a[5]  = (_Float16)r1[1]; a[6]  = (_Float16)r1[2]; a[7]  = (_Float16)r1[3];
  a[8]  = (_Float16)r2[0]; a[9]  = (_Float16)r2[1]; a[10] = (_Float16)r2[2]; a[11] = (_Float16)r2[3];
  a[12] = (_Float16)r3[0]; a[13] = (_Float16)r3[1]; a[14] = (_Float16)r3[2]; a[15] = (_Float16)r3[3];
  return a;
}

__global__ __launch_bounds__(256) void multiatt_kernel(
    const float* __restrict__ relation,   // (B,HOP,64)
    const float* __restrict__ neighbor,   // (B,HOP,64)
    const float* __restrict__ w_t,        // (64,64)
    const float* __restrict__ w_r,        // (64,64)
    const float* __restrict__ v_a,        // (192,1)
    float* __restrict__ out,              // (B,HOP,1)
    int ngroups)
{
  // 16 B-tiles (mat:2 x nt:4 x kc:2), each: 32 lanes x 16 halves = 1 KB -> 16 KB
  // B layout (16-bit, 32x16): per lane, col N = lane&15, halves i=0..15 hold
  // K = kc*32 + (lane>>4)*8 + (i&7) + (i>>3)*16   (matches A table order).
  __shared__ v16h  ldsBt[16 * 32];
  __shared__ float ldsVa[128];

  _Float16* ldsB = (_Float16*)ldsBt;
  const int tid = threadIdx.x;

  for (int idx = tid; idx < 16 * 512; idx += 256) {
    const int tile = idx >> 9;
    const int rem  = idx & 511;
    const int ln   = rem >> 4;
    const int i    = rem & 15;
    const int mat  = tile >> 3;          // 0: w_r (relation), 1: w_t (neighbor)
    const int nt   = (tile >> 1) & 3;
    const int kc   = tile & 1;
    const int n    = nt * 16 + (ln & 15);
    const int k    = kc * 32 + (ln >> 4) * 8 + (i & 7) + (i >> 3) * 16;
    const float* w = mat ? w_t : w_r;
    ldsB[idx] = (_Float16)w[k * DIM + n];
  }
  for (int i = tid; i < 128; i += 256) ldsVa[i] = v_a[64 + i];  // skip zero head block
  __syncthreads();

  const int lane   = tid & 31;
  const int wave   = tid >> 5;
  const int half   = lane >> 4;
  const int lrow   = lane & 15;          // A row offset; also D column
  const int wtotal = gridDim.x * (blockDim.x >> 5);
  const int gwave  = blockIdx.x * (blockDim.x >> 5) + wave;

  for (int g = gwave; g < ngroups; g += wtotal) {
    // Prefetch next grid-stride group (lane-spread across the 2x8KB regions).
    const int gn = g + wtotal;
    if (gn < ngroups) {
      const size_t nb = (size_t)gn * HOPN * DIM;
      __builtin_prefetch(relation + nb + (size_t)lane * 64, 0, 1);
      __builtin_prefetch(neighbor + nb + (size_t)lane * 64, 0, 1);
    }

    float s[2][8];
    #pragma unroll
    for (int mt = 0; mt < 2; ++mt)
      #pragma unroll
      for (int j = 0; j < 8; ++j) s[mt][j] = 0.0f;

    const size_t rowbase = (size_t)g * HOPN;

    #pragma unroll
    for (int mt = 0; mt < 2; ++mt) {
      const size_t row  = rowbase + mt * 16 + lrow;
      const float* prel = relation + row * DIM;
      const float* pnbr = neighbor + row * DIM;

      v16h Ar[2], An[2];
      #pragma unroll
      for (int kc = 0; kc < 2; ++kc) {
        const int kb = kc * 32 + half * 8;
        Ar[kc] = load_a_tile(prel, kb);
        An[kc] = load_a_tile(pnbr, kb);
      }

      #pragma unroll
      for (int nt = 0; nt < 4; ++nt) {
        v8f cr = {};
        v8f cn = {};
        #pragma unroll
        for (int kc = 0; kc < 2; ++kc) {
          v16h br = ldsBt[(0 * 8 + nt * 2 + kc) * 32 + lane];
          v16h bn = ldsBt[(1 * 8 + nt * 2 + kc) * 32 + lane];
          cr = __builtin_amdgcn_wmma_f32_16x16x32_f16(false, Ar[kc], false, br,
                                                      (short)0, cr, false, false);
          cn = __builtin_amdgcn_wmma_f32_16x16x32_f16(false, An[kc], false, bn,
                                                      (short)0, cn, false, false);
        }
        // D layout: this lane holds rows (half*8 + j), column (nt*16 + lrow)
        const float var = ldsVa[nt * 16 + lrow];        // v_a[64  + col]
        const float van = ldsVa[64 + nt * 16 + lrow];   // v_a[128 + col]
        #pragma unroll
        for (int j = 0; j < 8; ++j)
          s[mt][j] += fast_tanh(cr[j]) * var + fast_tanh(cn[j]) * van;
      }
    }

    // Reduce scores over the 16 columns (lanes within each half).
    #pragma unroll
    for (int mt = 0; mt < 2; ++mt)
      #pragma unroll
      for (int j = 0; j < 8; ++j) {
        float v = s[mt][j];
        v += __shfl_xor(v, 1, 32);
        v += __shfl_xor(v, 2, 32);
        v += __shfl_xor(v, 4, 32);
        v += __shfl_xor(v, 8, 32);
        s[mt][j] = v;   // rows mt*16 + half*8 + j, replicated in 16 lanes
      }

    // Softmax over 32 hops: each lane holds 16 scores; exchange with half^1.
    float m = -INFINITY;
    #pragma unroll
    for (int mt = 0; mt < 2; ++mt)
      #pragma unroll
      for (int j = 0; j < 8; ++j) m = fmaxf(m, s[mt][j]);
    m = fmaxf(m, __shfl_xor(m, 16, 32));

    float esum = 0.0f;
    #pragma unroll
    for (int mt = 0; mt < 2; ++mt)
      #pragma unroll
      for (int j = 0; j < 8; ++j) {
        s[mt][j] = fast_exp_nonpos(s[mt][j] - m);
        esum += s[mt][j];
      }
    esum += __shfl_xor(esum, 16, 32);
    const float inv = __builtin_amdgcn_rcpf(esum);

    if (lrow == 0) {  // lane 0 -> rows 0-7 & 16-23, lane 16 -> rows 8-15 & 24-31
      float* o = out + rowbase;
      #pragma unroll
      for (int mt = 0; mt < 2; ++mt)
        #pragma unroll
        for (int j = 0; j < 8; ++j)
          o[mt * 16 + half * 8 + j] = s[mt][j] * inv;
    }
  }
}

extern "C" void kernel_launch(void* const* d_in, const int* in_sizes, int n_in,
                              void* d_out, int out_size, void* d_ws, size_t ws_size,
                              hipStream_t stream) {
  // inputs: 0 entity(unused), 1 relation, 2 neighbor, 3 w_h(unused), 4 w_t, 5 w_r, 6 v_a
  const float* relation = (const float*)d_in[1];
  const float* neighbor = (const float*)d_in[2];
  const float* w_t      = (const float*)d_in[4];
  const float* w_r      = (const float*)d_in[5];
  const float* v_a      = (const float*)d_in[6];
  float* out = (float*)d_out;

  const int ngroups = in_sizes[1] / (HOPN * DIM);   // B
  const int blocks  = 1024;                         // persistent, grid-stride over b-groups
  multiatt_kernel<<<blocks, 256, 0, stream>>>(relation, neighbor, w_t, w_r, v_a,
                                              out, ngroups);
}
